// LstmCrf_72670846649037
// MI455X (gfx1250) — matrix operand order repaired and checked by reading, not compiled, not run on hardware
//
#include <hip/hip_runtime.h>

typedef __attribute__((ext_vector_type(16))) _Float16 v16h;
typedef __attribute__((ext_vector_type(8)))  float    v8f;

#define NLAB     45
#define START_I  43
#define STOP_I   44
#define NEGINF  -10000.0f
#define LOG2E    1.44269504088896340736f
#define LN2      0.69314718055994530942f

// lane <-> lane^16 exchange as pure VALU (v_permlanex16_b32, identity selects)
__device__ __forceinline__ float xor16(float x) {
    int v = __builtin_bit_cast(int, x);
    v = __builtin_amdgcn_permlanex16(v, v, 0x76543210, 0xfedcba98, false, false);
    return __builtin_bit_cast(float, v);
}
// raw hardware transcendentals (single v_exp_f32 / v_log_f32)
__device__ __forceinline__ float fexp(float x) { return __builtin_amdgcn_exp2f(x * LOG2E); }
__device__ __forceinline__ float flog(float x) { return __builtin_amdgcn_logf(x) * LN2; }

// One wave processes 16 batch sequences (one per WMMA column).
// alpha (48x16, padded from 45) lives in C/D fragment registers:
//   lane l, frag f, elem r -> row = 16*f + r + 8*(l>=16), col = l%16.
// E = exp(transitions), zero-padded, lives in A fragment registers.
// Pad rows need NO guards: E pad rows are exactly 0 -> WMMA d = 0 ->
// log(0) = -inf -> pad alpha self-maintains at -inf through max/exp/LSE.
__global__ __launch_bounds__(32)
void crf_fwd_wmma(const float* __restrict__ logits,
                  const int*   __restrict__ lens,
                  const float* __restrict__ trans,
                  float*       __restrict__ out,
                  int B, int S)
{
    const int lane = threadIdx.x & 31;
    const int col  = lane & 15;
    const int hi   = lane >> 4;
    const int gidx = blockIdx.x * 16 + col;
    const int batch = (gidx < B) ? gidx : (B - 1);

    // ---- E fragments (A layout, f16): lanes 0-15 K=0..15, lanes 16-31 K=16..31
    v16h Afrag[3][2];
    const int koff = hi ? 16 : 0;
    for (int mt = 0; mt < 3; ++mt) {
        const int i = mt * 16 + col;
        for (int kt = 0; kt < 2; ++kt) {
            v16h a;
            for (int e = 0; e < 16; ++e) {
                int k = koff + e;
                int j = kt * 32 + k;
                float v = (i < NLAB && j < NLAB) ? fexp(trans[i * NLAB + j]) : 0.0f;
                a[e] = (_Float16)v;
            }
            Afrag[mt][kt] = a;
        }
    }

    const int mylen = lens[batch];

    v8f alpha[3];
    for (int f = 0; f < 3; ++f)
        for (int r = 0; r < 8; ++r) {
            int row = 16 * f + r + 8 * hi;
            alpha[f][r] = (row == START_I) ? 0.0f : NEGINF;
        }

    const float* lbase = logits + (size_t)batch * S * NLAB;

    for (int t = 0; t < S; ++t) {
        // per-column max across 48 rows (pads are <= -10000, never win)
        float m = alpha[0][0];
        for (int r = 1; r < 8; ++r) m = fmaxf(m, alpha[0][r]);
        for (int r = 0; r < 8; ++r) m = fmaxf(m, alpha[1][r]);
        for (int r = 0; r < 8; ++r) m = fmaxf(m, alpha[2][r]);
        m = fmaxf(m, xor16(m));

        // partner exchange (only the halves each side needs): 16 permutes
        float q0[8], q1[8];
        for (int r = 0; r < 8; ++r) {
            q0[r] = xor16(hi ? alpha[0][r] : alpha[1][r]); // lo gets partner f0, hi gets partner f1
            q1[r] = xor16(alpha[2][r]);                    // lo gets partner f2
        }

        // B fragments: lanes 0-15 own K=base+0..15, lanes 16-31 K=base+16..31
        v16h B0, B1;
        for (int e = 0; e < 16; ++e) {
            float v0 = (hi == 0) ? ((e < 8) ? alpha[0][e] : q0[e - 8])        // rows 0..15
                                 : ((e < 8) ? q0[e]       : alpha[1][e - 8]); // rows 16..31
            float v1 = (hi == 0) ? ((e < 8) ? alpha[2][e] : q1[e - 8])        // rows 32..47
                                 : NEGINF;                                     // rows 48..63 pad
            B0[e] = (_Float16)fexp(v0 - m);
            B1[e] = (_Float16)fexp(v1 - m);
        }

        // D = E * a : 3 row tiles x 2 K chunks
        v8f d[3];
        for (int mt = 0; mt < 3; ++mt) {
            v8f c = {};
            c = __builtin_amdgcn_wmma_f32_16x16x32_f16(
                    false, Afrag[mt][0], false, B0, (short)0, c, false, false);
            c = __builtin_amdgcn_wmma_f32_16x16x32_f16(
                    false, Afrag[mt][1], false, B1, (short)0, c, false, false);
            d[mt] = c;
        }

        // alpha'[i] = logit[i] + m + log(D[i]) -- completely branch-free:
        // pad rows get log(0) = -inf and stay -inf forever (correctly ignored).
        const float* lt = lbase + (size_t)t * NLAB;
        const bool active = (t < mylen);
        for (int f = 0; f < 3; ++f) {
            const int rowbase = 16 * f + 8 * hi;
            for (int r = 0; r < 8; ++r) {
                int row  = rowbase + r;
                int rowc = (row < NLAB) ? row : (NLAB - 1);   // clamp: never OOB
                float nv = lt[rowc] + m + flog(d[f][r]);
                alpha[f][r] = active ? nv : alpha[f][r];
            }
        }
    }

    // ---- finalize: alpha += trans[STOP, j]; norm = LSE over rows per column ----
    // pad rows are <= -9998 or -inf: contribute exp(...) == 0, no guard needed.
    for (int f = 0; f < 3; ++f) {
        const int rowbase = 16 * f + 8 * hi;
        for (int r = 0; r < 8; ++r) {
            int row  = rowbase + r;
            int rowc = (row < NLAB) ? row : (NLAB - 1);
            alpha[f][r] += trans[STOP_I * NLAB + rowc];
        }
    }
    float m = alpha[0][0];
    for (int r = 1; r < 8; ++r) m = fmaxf(m, alpha[0][r]);
    for (int r = 0; r < 8; ++r) m = fmaxf(m, alpha[1][r]);
    for (int r = 0; r < 8; ++r) m = fmaxf(m, alpha[2][r]);
    m = fmaxf(m, xor16(m));
    float s = 0.0f;
    for (int f = 0; f < 3; ++f)
        for (int r = 0; r < 8; ++r) s += fexp(alpha[f][r] - m);
    s += xor16(s);
    float norm = m + flog(s);

    if (hi == 0 && gidx < B) out[gidx] = norm;
}

extern "C" void kernel_launch(void* const* d_in, const int* in_sizes, int n_in,
                              void* d_out, int out_size, void* d_ws, size_t ws_size,
                              hipStream_t stream) {
    const float* logits = (const float*)d_in[0];
    const int*   lens   = (const int*)d_in[1];
    const float* trans  = (const float*)d_in[2];
    float*       out    = (float*)d_out;

    const int B = in_sizes[1];
    const int S = in_sizes[0] / (B * NLAB);
    const int groups = (B + 15) / 16;

    crf_fwd_wmma<<<groups, 32, 0, stream>>>(logits, lens, trans, out, B, S);
}
